// DecoderGRU_46016279610175
// MI455X (gfx1250) — compile-verified, matrix operand order checked by reading, not run on hardware
//
#include <hip/hip_runtime.h>
#include <cstdint>
#include <cstddef>

typedef __bf16 bf16_t;
typedef __attribute__((ext_vector_type(16))) __bf16 v16bf;
typedef __attribute__((ext_vector_type(8)))  float  v8f;

#define B_   8
#define T_   128
#define V_   32000
#define E_   512
#define H_   1024
#define M_   512
#define S_   16
#define U_   128
#define W_   1024
#define CATK 2048   // H + 2M
#define GRU_BLOCKS 16

// ---------------- WMMA fragment helpers (ISA 7.12.2 layouts) ----------------
// A (16-bit, 16x32): lane L (M=L%16); halves: lanes<16 hold K={0..7,16..23},
// lanes>=16 hold K={8..15,24..31}  -> two contiguous 16B chunks per lane.
// B (32x16) symmetric with N on lanes (B loaded from row-major [N,K]).

__device__ __forceinline__ v16bf frag_from_bf16(const bf16_t* p, int ld, int row,
                                                int k, int lane) {
  int r  = row + (lane & 15);
  int kk = k + ((lane & 16) >> 1);            // +8 halves for upper half-wave
  const bf16_t* q = p + (size_t)r * ld + kk;
  union { float4 f[2]; v16bf v; } u;
  u.f[0] = *reinterpret_cast<const float4*>(q);
  u.f[1] = *reinterpret_cast<const float4*>(q + 16);
  return u.v;
}

__device__ __forceinline__ v8f wmma_bf16(v16bf a, v16bf b, v8f c) {
  return __builtin_amdgcn_wmma_f32_16x16x32_bf16(false, a, false, b, (short)0, c,
                                                 false, false);
}

__device__ __forceinline__ float sigmoidf_(float x) { return 1.0f / (1.0f + __expf(-x)); }

__device__ __forceinline__ void store_tile(v8f acc, int mbase, int n, int mh,
                                           float bs, float* Cf, int ldcf,
                                           bf16_t* Cb, int ldcb) {
#pragma unroll
  for (int v = 0; v < 8; ++v) {
    int m = mbase + v + mh;
    float val = acc[v] + bs;
    if (Cf) Cf[(size_t)m * ldcf + n] = val;
    if (Cb) Cb[(size_t)m * ldcb + n] = (bf16_t)val;
  }
}

// ---------------- WMMA GEMM:  C[M,N] = A[M,K] * B[N,K]^T + bias --------------
// A, B: bf16 row-major. Each wave computes a 64x16 strip (4 M-tiles) so one B
// fragment feeds 4 WMMAs (4x less B traffic). M%64==0, N%16==0, K%32==0.
// Consecutive waves share the same N-tile (B fragment reuse through L0/L2).
__global__ void __launch_bounds__(128)
k_gemm(const bf16_t* __restrict__ A, int lda,
       const bf16_t* __restrict__ Bw, int ldb,
       const float* __restrict__ bias,
       float* __restrict__ Cf, int ldcf,
       bf16_t* __restrict__ Cb, int ldcb,
       int M, int N, int K) {
  int wave = blockIdx.x * (blockDim.x >> 5) + (threadIdx.x >> 5);
  int lane = threadIdx.x & 31;
  int ntile = N >> 4;
  int mblk  = M >> 6;                       // 64-row strips
  int total = mblk * ntile;
  if (wave >= total) return;
  int mb = wave % mblk;                     // consecutive waves: same n-tile
  int nt = wave / mblk;
  int m0 = mb << 6;
  int n0 = nt << 4;
  v8f acc0 = {}, acc1 = {}, acc2 = {}, acc3 = {};
  for (int k = 0; k < K; k += 32) {
    v16bf b  = frag_from_bf16(Bw, ldb, n0, k, lane);
    v16bf a0 = frag_from_bf16(A, lda, m0,      k, lane);
    v16bf a1 = frag_from_bf16(A, lda, m0 + 16, k, lane);
    v16bf a2 = frag_from_bf16(A, lda, m0 + 32, k, lane);
    v16bf a3 = frag_from_bf16(A, lda, m0 + 48, k, lane);
    acc0 = wmma_bf16(a0, b, acc0);
    acc1 = wmma_bf16(a1, b, acc1);
    acc2 = wmma_bf16(a2, b, acc2);
    acc3 = wmma_bf16(a3, b, acc3);
  }
  int n   = n0 + (lane & 15);
  float bs = bias ? bias[n] : 0.0f;
  int mh  = (lane & 16) >> 1;               // +8 rows for upper half-wave
  store_tile(acc0, m0,      n, mh, bs, Cf, ldcf, Cb, ldcb);
  store_tile(acc1, m0 + 16, n, mh, bs, Cf, ldcf, Cb, ldcb);
  store_tile(acc2, m0 + 32, n, mh, bs, Cf, ldcf, Cb, ldcb);
  store_tile(acc3, m0 + 48, n, mh, bs, Cf, ldcf, Cb, ldcb);
}

// ---------------- persistent GRU layer --------------------------------------
// Agent-scope grid barrier across the 16 co-resident blocks.
__device__ __forceinline__ void grid_barrier(unsigned* ctr, unsigned target) {
  __threadfence();
  __syncthreads();
  if (threadIdx.x == 0) {
    __hip_atomic_fetch_add(ctr, 1u, __ATOMIC_ACQ_REL, __HIP_MEMORY_SCOPE_AGENT);
    while (__hip_atomic_load(ctr, __ATOMIC_ACQUIRE, __HIP_MEMORY_SCOPE_AGENT) < target)
      __builtin_amdgcn_s_sleep(2);
  }
  __syncthreads();
  __threadfence();
}

// One kernel runs ALL T timesteps of a GRU layer. 16 blocks x 128 thr = 64
// waves; wave handles 16 cols of H, computing r/z/n gate slices with WMMA.
// h ping-pong buffers are [16,H] (rows 8..15 stay zero = WMMA M padding),
// zero-initialized inside the kernel before the first barrier.
// W_hh is pre-converted bf16 (6.3 MB -> stays L2-resident across all steps).
__global__ void __launch_bounds__(128)
k_gru_layer(bf16_t* __restrict__ hA_b, float* __restrict__ hA_f,
            bf16_t* __restrict__ hB_b, float* __restrict__ hB_f,
            const bf16_t* __restrict__ Whh_b /*[3H,H] bf16*/,
            const float* __restrict__ bhh,
            const float* __restrict__ gi /*[B*T,3H] incl b_ih*/,
            bf16_t* __restrict__ hseq_b /*[B*T,H]*/,
            unsigned* __restrict__ ctr) {
  int wave = blockIdx.x * 4 + (threadIdx.x >> 5);
  int lane = threadIdx.x & 31;
  int n0 = wave << 4;

  // zero-init both ping-pong h buffers (all 2048 threads stride 16*H elems)
  {
    int gid = blockIdx.x * 128 + threadIdx.x;
    for (int i = gid; i < 16 * H_; i += GRU_BLOCKS * 128) {
      hA_f[i] = 0.0f; hB_f[i] = 0.0f;
      hA_b[i] = (bf16_t)0.0f; hB_b[i] = (bf16_t)0.0f;
    }
  }
  grid_barrier(ctr, GRU_BLOCKS);

  int n = n0 + (lane & 15);
  float bhr = bhh[n], bhz = bhh[H_ + n], bhn = bhh[2 * H_ + n];
  int mh = (lane & 16) >> 1;

  for (int t = 0; t < T_; ++t) {
    const bf16_t* hp_b = (t & 1) ? hB_b : hA_b;
    const float*  hp_f = (t & 1) ? hB_f : hA_f;
    bf16_t*       hn_b = (t & 1) ? hA_b : hB_b;
    float*        hn_f = (t & 1) ? hA_f : hB_f;

    v8f ar = {}, az = {}, an = {};
    for (int k = 0; k < H_; k += 32) {
      v16bf a  = frag_from_bf16(hp_b, H_, 0, k, lane);
      v16bf br = frag_from_bf16(Whh_b, H_, 0 * H_ + n0, k, lane);
      v16bf bz = frag_from_bf16(Whh_b, H_, 1 * H_ + n0, k, lane);
      v16bf bn = frag_from_bf16(Whh_b, H_, 2 * H_ + n0, k, lane);
      ar = wmma_bf16(a, br, ar);
      az = wmma_bf16(a, bz, az);
      an = wmma_bf16(a, bn, an);
    }
#pragma unroll
    for (int v = 0; v < 8; ++v) {
      int m = v + mh;                 // batch row
      if (m < B_) {
        const float* gir = gi + (size_t)(m * T_ + t) * (3 * H_);
        float r  = sigmoidf_(gir[n] + ar[v] + bhr);
        float z  = sigmoidf_(gir[H_ + n] + az[v] + bhz);
        float nn = tanhf(gir[2 * H_ + n] + r * (an[v] + bhn));
        float hp = hp_f[m * H_ + n];
        float hv = (1.0f - z) * nn + z * hp;
        hn_f[m * H_ + n] = hv;
        bf16_t hb = (bf16_t)hv;
        hn_b[m * H_ + n] = hb;
        hseq_b[(size_t)(m * T_ + t) * H_ + n] = hb;
      }
    }
    grid_barrier(ctr, (unsigned)(GRU_BLOCKS * (t + 2)));
  }
}

// ---------------- elementwise / reduction kernels ---------------------------
__global__ void k_gather_emb(const int* __restrict__ target,
                             const float* __restrict__ emb,
                             bf16_t* __restrict__ out, int n) {
  int i = blockIdx.x * blockDim.x + threadIdx.x;
  if (i >= n) return;
  int r = i / E_, e = i - r * E_;
  out[i] = (bf16_t)emb[(size_t)target[r] * E_ + e];
}

__global__ void k_f32_to_bf16(const float* __restrict__ x, bf16_t* __restrict__ y, int n) {
  int i = blockIdx.x * blockDim.x + threadIdx.x;
  if (i < n) y[i] = (bf16_t)x[i];
}

// wT[b][m][w] = (bf16) w_output[b][w][m]   (2M = W = 1024)
__global__ void k_transpose_w(const float* __restrict__ x, bf16_t* __restrict__ y, int n) {
  int i = blockIdx.x * blockDim.x + threadIdx.x;
  if (i >= n) return;
  int b = i >> 20;                  // / (1024*1024)
  int rem = i & ((1 << 20) - 1);
  int m = rem >> 10, wcol = rem & 1023;
  y[i] = (bf16_t)x[(size_t)(b * W_ + wcol) * 1024 + m];
}

__global__ void k_copy_rnn_to_cat(const bf16_t* __restrict__ rnn, bf16_t* __restrict__ cat, int n) {
  int i = blockIdx.x * blockDim.x + threadIdx.x;
  if (i >= n) return;
  int r = i >> 10, c = i & 1023;
  cat[(size_t)r * CATK + H_ + c] = rnn[i];
}

// masked softmax over last axis, in place; one block per (b,t) row. N <= 1024.
__global__ void __launch_bounds__(256)
k_masked_softmax(float* __restrict__ X, int N, const int* __restrict__ lens) {
  int r = blockIdx.x, b = r / T_, len = lens[b];
  float* row = X + (size_t)r * N;
  __shared__ float red[256];
  int tid = threadIdx.x;
  int cnt = (N + 255) >> 8;         // <= 4
  float vals[4];
  float mx = -3.0e38f;
  for (int j = 0; j < cnt; ++j) {
    int i = tid + (j << 8);
    float v = (i < N && i < len) ? row[i] : -3.0e38f;
    vals[j] = v; mx = fmaxf(mx, v);
  }
  red[tid] = mx; __syncthreads();
  for (int s = 128; s > 0; s >>= 1) { if (tid < s) red[tid] = fmaxf(red[tid], red[tid + s]); __syncthreads(); }
  mx = red[0]; __syncthreads();
  float sum = 0.0f;
  for (int j = 0; j < cnt; ++j) {
    int i = tid + (j << 8);
    float e = (i < N && i < len) ? __expf(vals[j] - mx) : 0.0f;
    vals[j] = e; sum += e;
  }
  red[tid] = sum; __syncthreads();
  for (int s = 128; s > 0; s >>= 1) { if (tid < s) red[tid] += red[tid + s]; __syncthreads(); }
  float inv = 1.0f / red[0];
  for (int j = 0; j < cnt; ++j) {
    int i = tid + (j << 8);
    if (i < N) row[i] = vals[j] * inv;
  }
}

// combine hierarchical scores + plain softmax over full W, write bf16 probs.
__global__ void __launch_bounds__(256)
k_combine_softmax(const float* __restrict__ ss, const float* __restrict__ su,
                  const float* __restrict__ sw, const int* __restrict__ seg_of_utt,
                  const int* __restrict__ utt_of_word, const int* __restrict__ w_len,
                  bf16_t* __restrict__ probs) {
  int r = blockIdx.x, b = r / T_;
  const float* ssr = ss + (size_t)r * S_;
  const float* sur = su + (size_t)r * U_;
  const float* swr = sw + (size_t)r * W_;
  int wl = w_len[b];
  __shared__ float red[256];
  int tid = threadIdx.x;
  float vals[4];
  float mx = -3.0e38f;
#pragma unroll
  for (int j = 0; j < 4; ++j) {
    int wcol = tid + (j << 8);
    float v = 0.0f;
    if (wcol < wl) {
      int u = utt_of_word[b * W_ + wcol];
      int s = seg_of_utt[b * U_ + u];
      v = ssr[s] * sur[u] * swr[wcol];
    }
    vals[j] = v; mx = fmaxf(mx, v);
  }
  red[tid] = mx; __syncthreads();
  for (int s = 128; s > 0; s >>= 1) { if (tid < s) red[tid] = fmaxf(red[tid], red[tid + s]); __syncthreads(); }
  mx = red[0]; __syncthreads();
  float sum = 0.0f;
#pragma unroll
  for (int j = 0; j < 4; ++j) { vals[j] = __expf(vals[j] - mx); sum += vals[j]; }
  red[tid] = sum; __syncthreads();
  for (int s = 128; s > 0; s >>= 1) { if (tid < s) red[tid] += red[tid + s]; __syncthreads(); }
  float inv = 1.0f / red[0];
#pragma unroll
  for (int j = 0; j < 4; ++j) probs[(size_t)r * W_ + tid + (j << 8)] = (bf16_t)(vals[j] * inv);
}

// in-place log_softmax per row, N = 32000
__global__ void __launch_bounds__(256)
k_log_softmax(float* __restrict__ X, int N) {
  int r = blockIdx.x;
  float* row = X + (size_t)r * N;
  __shared__ float red[256];
  int tid = threadIdx.x;
  float mx = -3.0e38f;
  for (int i = tid; i < N; i += 256) mx = fmaxf(mx, row[i]);
  red[tid] = mx; __syncthreads();
  for (int s = 128; s > 0; s >>= 1) { if (tid < s) red[tid] = fmaxf(red[tid], red[tid + s]); __syncthreads(); }
  mx = red[0]; __syncthreads();
  float sum = 0.0f;
  for (int i = tid; i < N; i += 256) sum += __expf(row[i] - mx);
  red[tid] = sum; __syncthreads();
  for (int s = 128; s > 0; s >>= 1) { if (tid < s) red[tid] += red[tid + s]; __syncthreads(); }
  float lse = mx + __logf(red[0]);
  __syncthreads();
  for (int i = tid; i < N; i += 256) row[i] = row[i] - lse;
}

// ---------------- host ------------------------------------------------------
static inline void launch_gemm(hipStream_t s, const bf16_t* A, int lda,
                               const bf16_t* Bw, int ldb, const float* bias,
                               float* Cf, int ldcf, bf16_t* Cb, int ldcb,
                               int M, int N, int K) {
  int waves = (M >> 6) * (N >> 4);
  int blocks = (waves + 3) >> 2;
  k_gemm<<<blocks, 128, 0, s>>>(A, lda, Bw, ldb, bias, Cf, ldcf, Cb, ldcb, M, N, K);
}

static inline void launch_conv(hipStream_t s, const float* x, bf16_t* y, int n) {
  k_f32_to_bf16<<<(n + 255) / 256, 256, 0, s>>>(x, y, n);
}

extern "C" void kernel_launch(void* const* d_in, const int* in_sizes, int n_in,
                              void* d_out, int out_size, void* d_ws, size_t ws_size,
                              hipStream_t stream) {
  (void)in_sizes; (void)n_in; (void)out_size; (void)ws_size;
  const int*   target    = (const int*)  d_in[0];
  const float* embedding = (const float*)d_in[1];
  const float* W_ih0     = (const float*)d_in[2];
  const float* W_hh0     = (const float*)d_in[3];
  const float* b_ih0     = (const float*)d_in[4];
  const float* b_hh0     = (const float*)d_in[5];
  const float* W_ih1     = (const float*)d_in[6];
  const float* W_hh1     = (const float*)d_in[7];
  const float* b_ih1     = (const float*)d_in[8];
  const float* b_hh1     = (const float*)d_in[9];
  const float* att_s_w   = (const float*)d_in[10];
  const float* att_s_b   = (const float*)d_in[11];
  const float* att_u_w   = (const float*)d_in[12];
  const float* att_u_b   = (const float*)d_in[13];
  const float* att_w_w   = (const float*)d_in[14];
  const float* att_w_b   = (const float*)d_in[15];
  const float* out_w     = (const float*)d_in[16];
  const float* out_b     = (const float*)d_in[17];
  const float* s_output  = (const float*)d_in[18];
  const float* u_output  = (const float*)d_in[19];
  const float* w_output  = (const float*)d_in[20];
  const int*   s_len     = (const int*)  d_in[21];
  const int*   u_len     = (const int*)  d_in[22];
  const int*   w_len     = (const int*)  d_in[23];
  const int*   seg_of_utt  = (const int*)d_in[24];
  const int*   utt_of_word = (const int*)d_in[25];
  float* out = (float*)d_out;

  char* wp = (char*)d_ws;
  auto alloc = [&](size_t bytes) -> void* {
    void* p = (void*)wp;
    wp += (bytes + 255) & ~(size_t)255;
    return p;
  };
  const int BT = B_ * T_;
  // activations
  bf16_t* emb_b   = (bf16_t*)alloc((size_t)BT * E_ * 2);
  float*  gi0     = (float*) alloc((size_t)BT * 3 * H_ * 4);
  float*  gi1     = (float*) alloc((size_t)BT * 3 * H_ * 4);
  float*  hA_f    = (float*) alloc(16 * H_ * 4);
  float*  hB_f    = (float*) alloc(16 * H_ * 4);
  bf16_t* hA_b    = (bf16_t*)alloc(16 * H_ * 2);
  bf16_t* hB_b    = (bf16_t*)alloc(16 * H_ * 2);
  unsigned* ctr   = (unsigned*)alloc(256);
  bf16_t* hseq0_b = (bf16_t*)alloc((size_t)BT * H_ * 2);
  bf16_t* rnn1_b  = (bf16_t*)alloc((size_t)BT * H_ * 2);
  bf16_t* sout_b  = (bf16_t*)alloc((size_t)B_ * S_ * M_ * 2);
  bf16_t* uout_b  = (bf16_t*)alloc((size_t)B_ * U_ * M_ * 2);
  bf16_t* wout_b  = (bf16_t*)alloc((size_t)B_ * W_ * 2 * M_ * 2);
  bf16_t* proj_s_b = (bf16_t*)alloc((size_t)B_ * S_ * H_ * 2);
  bf16_t* proj_u_b = (bf16_t*)alloc((size_t)B_ * U_ * H_ * 2);
  bf16_t* proj_w_b = (bf16_t*)alloc((size_t)B_ * W_ * H_ * 2);
  float*  sc_s    = (float*) alloc((size_t)BT * S_ * 4);
  float*  sc_u    = (float*) alloc((size_t)BT * U_ * 4);
  float*  sc_w    = (float*) alloc((size_t)BT * W_ * 4);
  bf16_t* wT_b    = (bf16_t*)alloc((size_t)B_ * 2 * M_ * W_ * 2);
  bf16_t* probs_b = (bf16_t*)alloc((size_t)BT * W_ * 2);
  bf16_t* cat_b   = (bf16_t*)alloc((size_t)BT * CATK * 2);
  // pre-converted bf16 weights (one-time; W_hh/out_w are the reuse-heavy ones)
  bf16_t* Wih0_b  = (bf16_t*)alloc((size_t)3 * H_ * E_ * 2);
  bf16_t* Whh0_b  = (bf16_t*)alloc((size_t)3 * H_ * H_ * 2);
  bf16_t* Wih1_b  = (bf16_t*)alloc((size_t)3 * H_ * H_ * 2);
  bf16_t* Whh1_b  = (bf16_t*)alloc((size_t)3 * H_ * H_ * 2);
  bf16_t* attsw_b = (bf16_t*)alloc((size_t)H_ * M_ * 2);
  bf16_t* attuw_b = (bf16_t*)alloc((size_t)H_ * M_ * 2);
  bf16_t* attww_b = (bf16_t*)alloc((size_t)H_ * 2 * M_ * 2);
  bf16_t* outw_b  = (bf16_t*)alloc((size_t)V_ * CATK * 2);   // 131 MB, fits L2

  // 0) one-time weight conversions fp32 -> bf16
  launch_conv(stream, W_ih0,   Wih0_b,  3 * H_ * E_);
  launch_conv(stream, W_hh0,   Whh0_b,  3 * H_ * H_);
  launch_conv(stream, W_ih1,   Wih1_b,  3 * H_ * H_);
  launch_conv(stream, W_hh1,   Whh1_b,  3 * H_ * H_);
  launch_conv(stream, att_s_w, attsw_b, H_ * M_);
  launch_conv(stream, att_u_w, attuw_b, H_ * M_);
  launch_conv(stream, att_w_w, attww_b, H_ * 2 * M_);
  launch_conv(stream, out_w,   outw_b,  V_ * CATK);

  // 1) embedding gather (fp32 -> bf16 rows), memory-side conversions
  {
    int n = BT * E_;
    k_gather_emb<<<(n + 255) / 256, 256, 0, stream>>>(target, embedding, emb_b, n);
  }
  launch_conv(stream, s_output, sout_b, B_ * S_ * M_);
  launch_conv(stream, u_output, uout_b, B_ * U_ * M_);
  launch_conv(stream, w_output, wout_b, B_ * W_ * 2 * M_);
  { int n = B_ * 2 * M_ * W_; k_transpose_w<<<(n + 255) / 256, 256, 0, stream>>>(w_output, wT_b, n); }

  // 2) gate-input GEMM for layer 0 over all timesteps
  launch_gemm(stream, emb_b, E_, Wih0_b, E_, b_ih0, gi0, 3 * H_, nullptr, 0, BT, 3 * H_, E_);

  // 3) GRU layer 0: single persistent kernel over all 128 timesteps
  hipMemsetAsync(ctr, 0, sizeof(unsigned), stream);
  k_gru_layer<<<GRU_BLOCKS, 128, 0, stream>>>(hA_b, hA_f, hB_b, hB_f,
                                              Whh0_b, b_hh0, gi0, hseq0_b, ctr);

  // 4) gate-input GEMM for layer 1, then persistent GRU layer 1
  launch_gemm(stream, hseq0_b, H_, Wih1_b, H_, b_ih1, gi1, 3 * H_, nullptr, 0, BT, 3 * H_, H_);
  hipMemsetAsync(ctr, 0, sizeof(unsigned), stream);
  k_gru_layer<<<GRU_BLOCKS, 128, 0, stream>>>(hA_b, hA_f, hB_b, hB_f,
                                              Whh1_b, b_hh1, gi1, rnn1_b, ctr);

  // 5) attention memory projections  proj = mem @ att_wᵀ + b  (bf16 out only)
  launch_gemm(stream, sout_b, M_, attsw_b, M_, att_s_b, nullptr, 0, proj_s_b, H_, B_ * S_, H_, M_);
  launch_gemm(stream, uout_b, M_, attuw_b, M_, att_u_b, nullptr, 0, proj_u_b, H_, B_ * U_, H_, M_);
  launch_gemm(stream, wout_b, 2 * M_, attww_b, 2 * M_, att_w_b, nullptr, 0, proj_w_b, H_, B_ * W_, H_, 2 * M_);

  // 6) attention scores per batch: scores[t,n] = q[t,:]·proj[n,:]
  for (int b = 0; b < B_; ++b) {
    launch_gemm(stream, rnn1_b + (size_t)b * T_ * H_, H_, proj_s_b + (size_t)b * S_ * H_, H_,
                nullptr, sc_s + (size_t)b * T_ * S_, S_, nullptr, 0, T_, S_, H_);
    launch_gemm(stream, rnn1_b + (size_t)b * T_ * H_, H_, proj_u_b + (size_t)b * U_ * H_, H_,
                nullptr, sc_u + (size_t)b * T_ * U_, U_, nullptr, 0, T_, U_, H_);
    launch_gemm(stream, rnn1_b + (size_t)b * T_ * H_, H_, proj_w_b + (size_t)b * W_ * H_, H_,
                nullptr, sc_w + (size_t)b * T_ * W_, W_, nullptr, 0, T_, W_, H_);
  }

  // 7) masked softmaxes (in place)
  k_masked_softmax<<<BT, 256, 0, stream>>>(sc_s, S_, s_len);
  k_masked_softmax<<<BT, 256, 0, stream>>>(sc_u, U_, u_len);
  k_masked_softmax<<<BT, 256, 0, stream>>>(sc_w, W_, w_len);

  // 8) hierarchical combine + softmax over full W -> bf16 probs
  k_combine_softmax<<<BT, 256, 0, stream>>>(sc_s, sc_u, sc_w, seg_of_utt, utt_of_word,
                                            w_len, probs_b);

  // 9) context per batch: cat[:, 0:2M] = probs @ w_output   (B operand = wT)
  for (int b = 0; b < B_; ++b) {
    launch_gemm(stream, probs_b + (size_t)b * T_ * W_, W_, wT_b + (size_t)b * 2 * M_ * W_, W_,
                nullptr, nullptr, 0, cat_b + (size_t)b * T_ * CATK, CATK, T_, 2 * M_, W_);
  }
  // 10) cat[:, 2M:2M+H] = rnn_out
  {
    int n = BT * H_;
    k_copy_rnn_to_cat<<<(n + 255) / 256, 256, 0, stream>>>(rnn1_b, cat_b, n);
  }

  // 11) output projection (dominant GEMM, 134 GFLOP) -> logits in d_out
  launch_gemm(stream, cat_b, CATK, outw_b, CATK, out_b, out, V_, nullptr, 0, BT, V_, CATK);

  // 12) log-softmax in place on d_out
  k_log_softmax<<<BT, 256, 0, stream>>>(out, V_);
}
